// FocusedLinearAttention_42305427866176
// MI455X (gfx1250) — matrix operand order repaired and checked by reading, not compiled, not run on hardware
//
#include <hip/hip_runtime.h>
#include <hip/hip_bf16.h>
#include <stdint.h>

#define NBATCH 8
#define LSEQ   4096
#define NHEAD  8
#define DDIM   64
#define EPSV   1e-6f

typedef __bf16 bf16_t;
typedef __attribute__((ext_vector_type(16))) __bf16 v16bf;
typedef __attribute__((ext_vector_type(8)))  float  v8f;

union BFrag {
    v16bf v;
    uint4 q[2];
    unsigned short s[16];
};

__device__ __forceinline__ unsigned short f2bf(float f) {
    union { float f; uint32_t u; } c; c.f = f;
    uint32_t u = c.u;
    uint32_t r = u + 0x7fffu + ((u >> 16) & 1u);   // round-to-nearest-even
    return (unsigned short)(r >> 16);
}

__device__ __forceinline__ float bf2f(unsigned short b) {
    union { uint32_t u; float f; } c; c.u = ((uint32_t)b) << 16;
    return c.f;
}

__device__ __forceinline__ float waveRedSum(float x) {
#pragma unroll
    for (int m = 16; m >= 1; m >>= 1) x += __shfl_xor(x, m, 32);
    return x;
}

// Async copy global -> LDS (16B per lane), tracked by ASYNCcnt.
__device__ __forceinline__ void async_load_b128(uint32_t lds_addr, const void* gptr) {
    asm volatile("global_load_async_to_lds_b128 %0, %1, off"
                 :: "v"(lds_addr), "v"(gptr) : "memory");
}

// ---------------------------------------------------------------- zero ksum
__global__ void k_zero(float* p, int n) {
    int i = blockIdx.x * blockDim.x + threadIdx.x;
    if (i < n) p[i] = 0.f;
}

// ------------------------------------------------- featurize q,k; stage bf16
// block = (n*H + h)*64 + lblk ; 256 threads = 8 waves, wave handles 8 rows,
// lane handles d = 2*lane, 2*lane+1. Writes:
//   qf  [n,l,h,d]  bf16 (natural, A-operand of O-GEMM)
//   kT  [n,h,d,l]  bf16 (A-operand of KV-GEMM, via LDS transpose)
//   vT  [n,h,v,l]  bf16 (B-operand of KV-GEMM, via LDS transpose)
//   ksum[n,h,d]    f32 (global atomics, one per block per d)
__global__ __launch_bounds__(256) void k_featurize(
    const float* __restrict__ q, const float* __restrict__ k,
    const float* __restrict__ v, const float* __restrict__ scale_p,
    unsigned short* __restrict__ qf, unsigned short* __restrict__ kT,
    unsigned short* __restrict__ vT, float* __restrict__ ksum)
{
    __shared__ unsigned short kT_lds[DDIM * 64];
    __shared__ unsigned short vT_lds[DDIM * 64];
    __shared__ float ksum_lds[DDIM];

    const int bid  = blockIdx.x;
    const int lblk = bid & 63;
    const int nh   = bid >> 6;
    const int h    = nh & (NHEAD - 1);
    const int n    = nh / NHEAD;
    const int l0   = lblk * 64;

    const int tid  = threadIdx.x;
    const int wave = tid >> 5;
    const int lane = tid & 31;
    const int d0   = lane * 2;

    if (tid < DDIM) ksum_lds[tid] = 0.f;
    __syncthreads();

    const float sc0 = log1pf(expf(scale_p[d0]));
    const float sc1 = log1pf(expf(scale_p[d0 + 1]));

    float ks0 = 0.f, ks1 = 0.f;

#pragma unroll
    for (int i = 0; i < 8; ++i) {
        const int l = l0 + wave * 8 + i;
        const size_t base = (((size_t)n * LSEQ + l) * NHEAD + h) * DDIM + d0;
        const float2 qv = *(const float2*)(q + base);
        const float2 kv = *(const float2*)(k + base);
        const float2 vv = *(const float2*)(v + base);

        // q path: relu+eps, /scale, focusing power 3, rescale to original norm
        float q0 = (fmaxf(qv.x, 0.f) + 1e-6f) / sc0;
        float q1 = (fmaxf(qv.y, 0.f) + 1e-6f) / sc1;
        float n1 = waveRedSum(q0 * q0 + q1 * q1);
        float q30 = q0 * q0 * q0, q31 = q1 * q1 * q1;
        float n3 = waveRedSum(q30 * q30 + q31 * q31);
        float fq = sqrtf(n1 / fmaxf(n3, 1e-30f));
        q30 *= fq; q31 *= fq;

        // k path
        float k0 = (fmaxf(kv.x, 0.f) + 1e-6f) / sc0;
        float k1 = (fmaxf(kv.y, 0.f) + 1e-6f) / sc1;
        float m1 = waveRedSum(k0 * k0 + k1 * k1);
        float k30 = k0 * k0 * k0, k31 = k1 * k1 * k1;
        float m3 = waveRedSum(k30 * k30 + k31 * k31);
        float fk = sqrtf(m1 / fmaxf(m3, 1e-30f));
        k30 *= fk; k31 *= fk;

        ks0 += k30; ks1 += k31;

        const unsigned int qp =
            (unsigned int)f2bf(q30) | ((unsigned int)f2bf(q31) << 16);
        *(unsigned int*)(qf + base) = qp;

        const int ll = l - l0;
        kT_lds[d0 * 64 + ll]       = f2bf(k30);
        kT_lds[(d0 + 1) * 64 + ll] = f2bf(k31);
        vT_lds[d0 * 64 + ll]       = f2bf(vv.x);
        vT_lds[(d0 + 1) * 64 + ll] = f2bf(vv.y);
    }

    atomicAdd(&ksum_lds[d0], ks0);
    atomicAdd(&ksum_lds[d0 + 1], ks1);
    __syncthreads();

    // coalesced write-out of transposed tiles: thread -> (d, 16-elem quarter)
    const int d = tid >> 2, qtr = tid & 3;
    const size_t tb = (((size_t)nh) * DDIM + d) * LSEQ + l0 + qtr * 16;
    *(uint4*)(kT + tb) = *(const uint4*)(kT_lds + d * 64 + qtr * 16);
    *(uint4*)(vT + tb) = *(const uint4*)(vT_lds + d * 64 + qtr * 16);

    if (tid < DDIM) atomicAdd(&ksum[nh * DDIM + tid], ksum_lds[tid]);
}

// ------------------------------------------------------ z = 1/(q . sum_k + eps)
__global__ __launch_bounds__(256) void k_znorm(
    const unsigned short* __restrict__ qf, const float* __restrict__ ksum,
    float* __restrict__ z)
{
    const int row  = blockIdx.x * 8 + (threadIdx.x >> 5); // over N*L*H
    const int lane = threadIdx.x & 31;
    const int h  = row & (NHEAD - 1);
    const int nl = row >> 3;              // n*L + l
    const int n  = nl >> 12;              // / LSEQ
    const int nh = n * NHEAD + h;

    const size_t base = (size_t)row * DDIM + lane * 2;
    const unsigned int qp = *(const unsigned int*)(qf + base);
    const float q0 = bf2f((unsigned short)(qp & 0xffffu));
    const float q1 = bf2f((unsigned short)(qp >> 16));
    const float s0 = ksum[nh * DDIM + lane * 2];
    const float s1 = ksum[nh * DDIM + lane * 2 + 1];
    const float dot = waveRedSum(q0 * s0 + q1 * s1);
    if (lane == 0) z[row] = 1.f / (dot + EPSV);
}

// --------------------------------------------- KV[d,v] = sum_s Kf[s,d] V[s,v]
// 64 blocks (one per n,h), 512 threads = 16 waves, wave -> (ti,tj) 16x16 tile.
// Double-buffered GLOBAL_LOAD_ASYNC_TO_LDS_B128 staging, 128 K-values/stage:
// each thread issues 4 async b128 per stage (tid<256 -> kT, tid>=256 -> vT);
// each wave runs 4 WMMAs (2 independent accumulator chains) per barrier pair.
// LDS rows padded to 136 ushorts: row base advances 4 banks/row, so the
// 16-lane x 16B fragment reads tile all 64 banks conflict-free.
#define KV_ROWP   136                      // padded row stride (ushorts)
#define KV_ARR    (DDIM * KV_ROWP)         // 8704 ushorts per array per buffer
#define KV_BUFB   (2 * KV_ARR * 2)         // 34816 bytes per buffer (k + v)

__global__ __launch_bounds__(512) void k_kv(
    const unsigned short* __restrict__ kT, const unsigned short* __restrict__ vT,
    unsigned short* __restrict__ kvT)     // [n*h, v, d] bf16 (B-operand of O-GEMM)
{
    __shared__ unsigned short stage[2][2 * KV_ARR];   // 68 KB

    const int nh   = blockIdx.x;
    const int tid  = threadIdx.x;
    const int wave = tid >> 5;
    const int lane = tid & 31;
    const int l15  = lane & 15;
    const int hiH  = lane >> 4;
    const int ti   = wave >> 2;
    const int tj   = wave & 3;

    // --- async-copy assignment: this thread's 4 x 16B slices of each stage
    const int sel  = tid >> 8;            // 0 = kT, 1 = vT
    const int t    = tid & 255;
    const int srow = t >> 2;              // source row (d or v), 0..63
    const int qtr  = t & 3;               // 8-elem offset within each 32-chunk
    const unsigned short* gsrc =
        (sel ? vT : kT) + (((size_t)nh) * DDIM + srow) * LSEQ + qtr * 8;
    const uint32_t ldsOff =
        (uint32_t)(uintptr_t)&stage[0][sel * KV_ARR + srow * KV_ROWP + qtr * 8];

    // --- fragment read offsets (LDS, padded rows)
    const int arow = (ti * 16 + l15) * KV_ROWP;            // A row (d)
    const int brow = (tj * 16 + l15) * KV_ROWP + KV_ARR;   // B row (v)
    const int a0 = hiH ? 8 : 0;           // A: K 0..7/16..23 vs 8..15/24..31
    const int b0 = hiH ? 8 : 0;           // B: K 0..7/16..23 vs 8..15/24..31

    const int nst = LSEQ / 128;           // 32 stages

#pragma unroll
    for (int j = 0; j < 4; ++j)           // prefetch stage 0 into buf 0
        async_load_b128(ldsOff + j * 64, gsrc + j * 32);

    v8f acc0 = {}, acc1 = {};
    for (int s = 0; s < nst; ++s) {
        const int buf = s & 1;
        if (s + 1 < nst) {
            const uint32_t dst = ldsOff + (buf ^ 1) * KV_BUFB;
            const unsigned short* g = gsrc + (size_t)(s + 1) * 128;
#pragma unroll
            for (int j = 0; j < 4; ++j)
                async_load_b128(dst + j * 64, g + j * 32);
            asm volatile("s_wait_asynccnt 0x4" ::: "memory"); // stage s done
        } else {
            asm volatile("s_wait_asynccnt 0x0" ::: "memory");
        }
        __syncthreads();                  // all waves' stage-s copies visible

        const unsigned short* sb = &stage[buf][0];
#pragma unroll
        for (int cc = 0; cc < 4; ++cc) {
            BFrag a, b;
            a.q[0] = *(const uint4*)(sb + arow + cc * 32 + a0);
            a.q[1] = *(const uint4*)(sb + arow + cc * 32 + a0 + 16);
            b.q[0] = *(const uint4*)(sb + brow + cc * 32 + b0);
            b.q[1] = *(const uint4*)(sb + brow + cc * 32 + b0 + 16);
            if (cc & 1)
                acc1 = __builtin_amdgcn_wmma_f32_16x16x32_bf16(
                    false, a.v, false, b.v, (short)0, acc1, false, false);
            else
                acc0 = __builtin_amdgcn_wmma_f32_16x16x32_bf16(
                    false, a.v, false, b.v, (short)0, acc0, false, false);
        }
        __syncthreads();                  // buf reusable for stage s+2 copy
    }

    const v8f acc = acc0 + acc1;

    // C layout: VGPR r, lanes<16 -> M=r, lanes>=16 -> M=r+8 ; N = l15
#pragma unroll
    for (int r = 0; r < 8; ++r) {
        const int m  = ti * 16 + r + (hiH ? 8 : 0);   // d index
        const int nn = tj * 16 + l15;                 // v index
        kvT[(((size_t)nh) * DDIM + nn) * DDIM + m] = f2bf(acc[r]);
    }
}

// ----- O[l,v] = (Qf @ KV)[l,v] * z[l] + dwconv5x5(values)[l,v]  (single store)
// grid (N*H, L/128); 256 threads = 8 waves; wave -> 16 rows x 64 cols.
__global__ __launch_bounds__(256) void k_out_fused(
    const unsigned short* __restrict__ qf, const unsigned short* __restrict__ kvT,
    const float* __restrict__ z, const float* __restrict__ vsrc,
    const float* __restrict__ w, const float* __restrict__ bias,
    float* __restrict__ out)
{
    __shared__ float wl[DDIM * 25];
    __shared__ float bl[DDIM];
    for (int i = threadIdx.x; i < DDIM * 25; i += 256) wl[i] = w[i];
    if (threadIdx.x < DDIM) bl[threadIdx.x] = bias[threadIdx.x];
    __syncthreads();

    const int nh   = blockIdx.x;
    const int lblk = blockIdx.y;
    const int h    = nh & (NHEAD - 1);
    const int n    = nh >> 3;
    const int tid  = threadIdx.x;
    const int wave = tid >> 5;
    const int lane = tid & 31;
    const int l15  = lane & 15;
    const int hiH  = lane >> 4;
    const int l0   = lblk * 128 + wave * 16;

    const unsigned short* aBase =
        qf + ((((size_t)n * LSEQ) + l0 + l15) * NHEAD + h) * DDIM;

    v8f acc[4] = {};
#pragma unroll
    for (int cc = 0; cc < 2; ++cc) {
        const int c = cc * 32;
        BFrag a;
        const int a0 = c + (hiH ? 8 : 0);
        a.q[0] = *(const uint4*)(aBase + a0);
        a.q[1] = *(const uint4*)(aBase + a0 + 16);
#pragma unroll
        for (int tj = 0; tj < 4; ++tj) {
            const unsigned short* bBase =
                kvT + (((size_t)nh) * DDIM + tj * 16 + l15) * DDIM;
            BFrag b;
            const int b0 = c + (hiH ? 8 : 0);
            b.q[0] = *(const uint4*)(bBase + b0);
            b.q[1] = *(const uint4*)(bBase + b0 + 16);
            acc[tj] = __builtin_amdgcn_wmma_f32_16x16x32_bf16(
                false, a.v, false, b.v, (short)0, acc[tj], false, false);
        }
    }

    float zr[8];
#pragma unroll
    for (int r = 0; r < 8; ++r) {
        const int l = l0 + r + (hiH ? 8 : 0);
        zr[r] = z[((size_t)n * LSEQ + l) * NHEAD + h];
    }

#pragma unroll
    for (int tj = 0; tj < 4; ++tj) {
        const int d = tj * 16 + l15;
#pragma unroll
        for (int r = 0; r < 8; ++r) {
            const int l = l0 + r + (hiH ? 8 : 0);
            const int y = l >> 6, x = l & 63;
            float o = acc[tj][r] * zr[r] + bl[d];
            for (int ky = 0; ky < 5; ++ky) {
                const int yy = y + ky - 2;
                if ((unsigned)yy > 63u) continue;
                for (int kx = 0; kx < 5; ++kx) {
                    const int xx = x + kx - 2;
                    if ((unsigned)xx > 63u) continue;
                    o += wl[d * 25 + ky * 5 + kx] *
                         vsrc[((((size_t)n * LSEQ) + (yy << 6) + xx) * NHEAD + h)
                              * DDIM + d];
                }
            }
            out[(((size_t)n * LSEQ + l) * NHEAD + h) * DDIM + d] = o;
        }
    }
}

extern "C" void kernel_launch(void* const* d_in, const int* in_sizes, int n_in,
                              void* d_out, int out_size, void* d_ws, size_t ws_size,
                              hipStream_t stream) {
    (void)in_sizes; (void)n_in; (void)out_size; (void)ws_size;
    const float* q       = (const float*)d_in[0];
    const float* k       = (const float*)d_in[1];
    const float* v       = (const float*)d_in[2];
    const float* scale_p = (const float*)d_in[3];
    const float* dwc_w   = (const float*)d_in[4];
    const float* dwc_b   = (const float*)d_in[5];
    // d_in[6] = height (==64), baked into the index math.

    char* ws = (char*)d_ws;
    const size_t MB = 1024 * 1024;
    unsigned short* qf  = (unsigned short*)(ws);                 // 32 MiB
    unsigned short* kT  = (unsigned short*)(ws + 32 * MB);       // 32 MiB
    unsigned short* vT  = (unsigned short*)(ws + 64 * MB);       // 32 MiB
    unsigned short* kvT = (unsigned short*)(ws + 96 * MB);       // 512 KiB
    float* ksum         = (float*)(ws + 97 * MB);                // 16 KiB
    float* z            = (float*)(ws + 98 * MB);                // 1 MiB
    float* out          = (float*)d_out;

    hipLaunchKernelGGL(k_zero, dim3(16), dim3(256), 0, stream,
                       ksum, NBATCH * NHEAD * DDIM);
    hipLaunchKernelGGL(k_featurize, dim3(NBATCH * NHEAD * (LSEQ / 64)), dim3(256),
                       0, stream, q, k, v, scale_p, qf, kT, vT, ksum);
    hipLaunchKernelGGL(k_znorm, dim3(NBATCH * LSEQ * NHEAD / 8), dim3(256),
                       0, stream, qf, ksum, z);
    hipLaunchKernelGGL(k_kv, dim3(NBATCH * NHEAD), dim3(512), 0, stream,
                       kT, vT, kvT);
    hipLaunchKernelGGL(k_out_fused, dim3(NBATCH * NHEAD, LSEQ / 128), dim3(256),
                       0, stream, qf, kvT, z, v, dwc_w, dwc_b, out);
}
